// SimpleNSVQ_78632261255354
// MI455X (gfx1250) — matrix operand order repaired and checked by reading, not compiled, not run on hardware
//
#include <hip/hip_runtime.h>
#include <hip/hip_bf16.h>
#include <math.h>

// Problem constants (from reference): B=8, T=2048, D=256, K=8192
#define D_DIM   256
#define K_CODES 8192
#define N_ROWS  16384
#define TILE_K  16            // codes per LDS tile
#define NTILES  (K_CODES / TILE_K)
#define CB_PAD  8             // bf16 pad -> row stride 264 (528B = 132 dwords, bank-stride 4, conflict-free)
#define CB_STRIDE (D_DIM + CB_PAD)

typedef __attribute__((ext_vector_type(16))) __bf16 v16bf;
typedef __attribute__((ext_vector_type(8)))  float  v8f;

// ---------------------------------------------------------------------------
// Kernel A: e_sq[k] = sum_d codebook[k][d]^2   (one wave per code row)
//           also zeroes the loss accumulator for this call.
// ---------------------------------------------------------------------------
__global__ void __launch_bounds__(256) nsvq_esq_kernel(const float* __restrict__ cb,
                                                       float* __restrict__ esq,
                                                       float* __restrict__ lossAccum) {
    const int tid  = threadIdx.x;
    const int lane = tid & 31;
    const int wave = tid >> 5;
    const int row  = blockIdx.x * 8 + wave;           // grid = K/8 = 1024 blocks

    const float* p = cb + (size_t)row * D_DIM + lane * 8;
    const float4 a = *reinterpret_cast<const float4*>(p);
    const float4 b = *reinterpret_cast<const float4*>(p + 4);
    float s = a.x*a.x + a.y*a.y + a.z*a.z + a.w*a.w
            + b.x*b.x + b.y*b.y + b.z*b.z + b.w*b.w;
    #pragma unroll
    for (int off = 1; off < 32; off <<= 1) s += __shfl_xor(s, off, 32);
    if (lane == 0) esq[row] = s;
    if (blockIdx.x == 0 && tid == 0) lossAccum[0] = 0.f;
}

// ---------------------------------------------------------------------------
// Kernel B: fused bf16-WMMA GEMM + argmin.
//   argmin_k ( e_sq[k] - 2 * dot(x_row, code_k) )   (x_sq is row-constant)
// Block = 256 threads = 8 waves; each wave owns 16 rows (A fragments held in
// registers, all K=256). Codebook streamed through a double-buffered LDS tile
// (16 codes) with register-staged global loads overlapping the WMMA chain.
// TWO independent accumulator chains (even/odd K-chunks) so ds_load waits of
// one chain overlap WMMA issue of the other.
// ---------------------------------------------------------------------------
__global__ void __launch_bounds__(256) nsvq_argmin_kernel(const float* __restrict__ x,
                                                          const float* __restrict__ codebook,
                                                          const float* __restrict__ esq,
                                                          int* __restrict__ idx_out) {
    __shared__ __align__(16) __bf16 cbTile[2][TILE_K][CB_STRIDE];

    const int tid  = threadIdx.x;
    const int lane = tid & 31;
    const int wave = tid >> 5;
    const int half = (lane >> 4) & 1;   // 0: lanes 0-15, 1: lanes 16-31
    const int l15  = lane & 15;

    const int m0 = blockIdx.x * 128 + wave * 16;      // grid = N/128 = 128 blocks

    // ---- Preload A fragments: rows m0..m0+15, all 256 K, as bf16 WMMA layout.
    // 16x32 bf16 A: lane<16 -> K = kb..kb+15; lane>=16 -> +16.
    v16bf a_frag[8];
    {
        const float* xrow = x + (size_t)(m0 + l15) * D_DIM;
        #pragma unroll
        for (int c = 0; c < 8; ++c) {
            const int kb = c * 32 + half * 16;
            const float4 s0 = *reinterpret_cast<const float4*>(xrow + kb);
            const float4 s1 = *reinterpret_cast<const float4*>(xrow + kb + 4);
            const float4 s2 = *reinterpret_cast<const float4*>(xrow + kb + 8);
            const float4 s3 = *reinterpret_cast<const float4*>(xrow + kb + 12);
            v16bf a;
            a[0]  = (__bf16)s0.x; a[1]  = (__bf16)s0.y; a[2]  = (__bf16)s0.z; a[3]  = (__bf16)s0.w;
            a[4]  = (__bf16)s1.x; a[5]  = (__bf16)s1.y; a[6]  = (__bf16)s1.z; a[7]  = (__bf16)s1.w;
            a[8]  = (__bf16)s2.x; a[9]  = (__bf16)s2.y; a[10] = (__bf16)s2.z; a[11] = (__bf16)s2.w;
            a[12] = (__bf16)s3.x; a[13] = (__bf16)s3.y; a[14] = (__bf16)s3.z; a[15] = (__bf16)s3.w;
            a_frag[c] = a;
        }
    }

    float minv[8];
    int   mini[8];
    #pragma unroll
    for (int r = 0; r < 8; ++r) { minv[r] = 3.0e38f; mini[r] = 0; }

    // Cooperative tile-load mapping: 256 threads x 16 floats = 16x256 tile.
    const int ldRow = tid >> 4;          // 0..15 (code within tile)
    const int ldCol = (tid & 15) * 16;   // 0..240 (k segment)
    const float* ldBase = codebook + (size_t)ldRow * D_DIM + ldCol;

    // Load 16 f32 of tile `code0` -> 16 bf16 packed in two uint4 registers.
    auto loadStage = [&](int code0, uint4& q0, uint4& q1) {
        const float* src = ldBase + (size_t)code0 * D_DIM;
        const float4 f0 = *reinterpret_cast<const float4*>(src);
        const float4 f1 = *reinterpret_cast<const float4*>(src + 4);
        const float4 f2 = *reinterpret_cast<const float4*>(src + 8);
        const float4 f3 = *reinterpret_cast<const float4*>(src + 12);
        union { __bf16 h[16]; uint4 q[2]; } u;
        u.h[0]  = (__bf16)f0.x; u.h[1]  = (__bf16)f0.y; u.h[2]  = (__bf16)f0.z; u.h[3]  = (__bf16)f0.w;
        u.h[4]  = (__bf16)f1.x; u.h[5]  = (__bf16)f1.y; u.h[6]  = (__bf16)f1.z; u.h[7]  = (__bf16)f1.w;
        u.h[8]  = (__bf16)f2.x; u.h[9]  = (__bf16)f2.y; u.h[10] = (__bf16)f2.z; u.h[11] = (__bf16)f2.w;
        u.h[12] = (__bf16)f3.x; u.h[13] = (__bf16)f3.y; u.h[14] = (__bf16)f3.z; u.h[15] = (__bf16)f3.w;
        q0 = u.q[0];
        q1 = u.q[1];
    };
    auto storeStage = [&](int buf, uint4 q0, uint4 q1) {
        uint4* dst = reinterpret_cast<uint4*>(&cbTile[buf][ldRow][ldCol]);
        dst[0] = q0;
        dst[1] = q1;
    };

    // Prolog: stage tile 0 into buffer 0.
    {
        uint4 p0, p1;
        loadStage(0, p0, p1);
        storeStage(0, p0, p1);
    }

    for (int t = 0; t < NTILES; ++t) {
        const int buf   = t & 1;
        const int code0 = t * TILE_K;
        const bool more = (t + 1 < NTILES);

        __syncthreads();   // tile t visible; everyone done reading buf^1 (tile t-1)

        // Issue next tile's global loads early; latency hides under the WMMAs.
        uint4 n0, n1;
        if (more) {
            if (t + 2 < NTILES) {
                __builtin_prefetch(ldBase + (size_t)(code0 + 2 * TILE_K) * D_DIM, 0, 0);
            }
            loadStage(code0 + TILE_K, n0, n1);
        }

        // e_sq for this lane's column (L2-resident, issued early).
        const float es  = esq[code0 + l15];
        const int   col = code0 + l15;

        // ---- Two independent WMMA chains over K=256 (even chunks -> acc0,
        // odd chunks -> acc1) so the scheduler overlaps ds waits with WMMAs.
        // 32x16 bf16 B: lane<16 -> K=c*32+0..15, lane>=16 -> K=c*32+16..31, col=l15.
        const __bf16* bbase = &cbTile[buf][l15][half * 16];
        v8f acc0 = {};
        v8f acc1 = {};
        #pragma unroll
        for (int p = 0; p < 4; ++p) {
            union { uint4 q[2]; v16bf v; } b0, b1;
            const uint4* bp0 = reinterpret_cast<const uint4*>(bbase + (2 * p) * 32);
            const uint4* bp1 = reinterpret_cast<const uint4*>(bbase + (2 * p + 1) * 32);
            b0.q[0] = bp0[0]; b0.q[1] = bp0[1];
            b1.q[0] = bp1[0]; b1.q[1] = bp1[1];
            acc0 = __builtin_amdgcn_wmma_f32_16x16x32_bf16(
                false, a_frag[2 * p],     false, b0.v, (short)0, acc0, false, false);
            acc1 = __builtin_amdgcn_wmma_f32_16x16x32_bf16(
                false, a_frag[2 * p + 1], false, b1.v, (short)0, acc1, false, false);
        }

        // ---- running argmin on the accumulator fragments.
        // C layout: VGPR r -> row m0 + half*8 + r; lane -> column code0 + l15.
        #pragma unroll
        for (int r = 0; r < 8; ++r) {
            const float d = fmaf(-2.0f, acc1[r], fmaf(-2.0f, acc0[r], es));
            if (d < minv[r]) { minv[r] = d; mini[r] = col; }
        }

        // Store staged tile t+1 into the alternate buffer (safe: barrier above
        // guaranteed all waves finished reading it as tile t-1).
        if (more) storeStage(buf ^ 1, n0, n1);
    }

    // ---- reduce across the 16 lanes sharing each row (xor<16 stays in-half).
    #pragma unroll
    for (int r = 0; r < 8; ++r) {
        float v = minv[r];
        int   i = mini[r];
        #pragma unroll
        for (int off = 1; off < 16; off <<= 1) {
            const float ov = __shfl_xor(v, off, 32);
            const int   oi = __shfl_xor(i, off, 32);
            if (ov < v || (ov == v && oi < i)) { v = ov; i = oi; }  // ties -> lowest index
        }
        if (l15 == 0) idx_out[m0 + half * 8 + r] = i;
    }
}

// ---------------------------------------------------------------------------
// Kernel C: per-row finalize (one wave per row).
//   resid_norm, noise_norm, quantized = x + (rn/(nn+eps))*noise,
//   indices (as float), accumulate sum of resid^2 for the loss.
// ---------------------------------------------------------------------------
__global__ void __launch_bounds__(256) nsvq_finalize_kernel(const float* __restrict__ x,
                                                            const float* __restrict__ noise,
                                                            const float* __restrict__ cb,
                                                            const int* __restrict__ idx,
                                                            float* __restrict__ qOut,
                                                            float* __restrict__ idxOut,
                                                            float* __restrict__ lossAccum) {
    const int tid  = threadIdx.x;
    const int lane = tid & 31;
    const int wave = tid >> 5;
    const int row  = blockIdx.x * 8 + wave;           // grid = N/8 = 2048 blocks

    const int code = idx[row];
    const size_t off = (size_t)row * D_DIM + lane * 8;
    const float* xr = x + off;
    const float* nr = noise + off;
    const float* cr = cb + (size_t)code * D_DIM + lane * 8;

    const float4 x0 = *reinterpret_cast<const float4*>(xr);
    const float4 x1 = *reinterpret_cast<const float4*>(xr + 4);
    const float4 n0 = *reinterpret_cast<const float4*>(nr);
    const float4 n1 = *reinterpret_cast<const float4*>(nr + 4);
    const float4 c0 = *reinterpret_cast<const float4*>(cr);
    const float4 c1 = *reinterpret_cast<const float4*>(cr + 4);

    float rsq = 0.f, nsq = 0.f;
    {
        float d;
        d = x0.x - c0.x; rsq += d * d;   d = x0.y - c0.y; rsq += d * d;
        d = x0.z - c0.z; rsq += d * d;   d = x0.w - c0.w; rsq += d * d;
        d = x1.x - c1.x; rsq += d * d;   d = x1.y - c1.y; rsq += d * d;
        d = x1.z - c1.z; rsq += d * d;   d = x1.w - c1.w; rsq += d * d;
        nsq += n0.x*n0.x + n0.y*n0.y + n0.z*n0.z + n0.w*n0.w;
        nsq += n1.x*n1.x + n1.y*n1.y + n1.z*n1.z + n1.w*n1.w;
    }
    #pragma unroll
    for (int o = 1; o < 32; o <<= 1) {
        rsq += __shfl_xor(rsq, o, 32);
        nsq += __shfl_xor(nsq, o, 32);
    }

    const float scale = sqrtf(rsq) / (sqrtf(nsq) + 1e-12f);

    float4 q0, q1;
    q0.x = x0.x + scale * n0.x; q0.y = x0.y + scale * n0.y;
    q0.z = x0.z + scale * n0.z; q0.w = x0.w + scale * n0.w;
    q1.x = x1.x + scale * n1.x; q1.y = x1.y + scale * n1.y;
    q1.z = x1.z + scale * n1.z; q1.w = x1.w + scale * n1.w;
    *reinterpret_cast<float4*>(qOut + off)     = q0;
    *reinterpret_cast<float4*>(qOut + off + 4) = q1;

    if (lane == 0) {
        idxOut[row] = (float)code;
        atomicAdd(lossAccum, rsq);
    }
}

// ---------------------------------------------------------------------------
// Kernel D: vq_loss = codebook_loss + 0.25*commitment = 1.25 * mean(resid^2)
// ---------------------------------------------------------------------------
__global__ void nsvq_loss_kernel(const float* __restrict__ lossAccum,
                                 float* __restrict__ lossOut) {
    lossOut[0] = 1.25f * lossAccum[0] / (float)((long long)N_ROWS * D_DIM);
}

// ---------------------------------------------------------------------------
extern "C" void kernel_launch(void* const* d_in, const int* in_sizes, int n_in,
                              void* d_out, int out_size, void* d_ws, size_t ws_size,
                              hipStream_t stream) {
    const float* x        = (const float*)d_in[0];  // [16384,256]
    const float* noise    = (const float*)d_in[1];  // [16384,256]
    const float* codebook = (const float*)d_in[2];  // [8192,256]

    float* out      = (float*)d_out;
    float* qOut     = out;                          // [16384*256]
    float* idxOut   = out + (size_t)N_ROWS * D_DIM; // [16384]
    float* lossOut  = idxOut + N_ROWS;              // [1]

    // Workspace layout: [0] loss accum (f32), [64..] e_sq (8192 f32), then idx (16384 i32)
    float* ws_f      = (float*)d_ws;
    float* lossAccum = ws_f;
    float* esq       = ws_f + 64;
    int*   idxBuf    = (int*)(ws_f + 64 + K_CODES);

    nsvq_esq_kernel<<<K_CODES / 8, 256, 0, stream>>>(codebook, esq, lossAccum);
    nsvq_argmin_kernel<<<N_ROWS / 128, 256, 0, stream>>>(x, codebook, esq, idxBuf);
    nsvq_finalize_kernel<<<N_ROWS / 8, 256, 0, stream>>>(x, noise, codebook, idxBuf,
                                                         qOut, idxOut, lossAccum);
    nsvq_loss_kernel<<<1, 1, 0, stream>>>(lossAccum, lossOut);
}
